// Re_gptEncoder_85907935855134
// MI455X (gfx1250) — matrix operand order repaired and checked by reading, not compiled, not run on hardware
//
#include <hip/hip_runtime.h>
#include <hip/hip_bf16.h>
#include <math.h>

// ---------------------------------------------------------------------------
// Types for CDNA5 WMMA (wave32)
// ---------------------------------------------------------------------------
typedef __attribute__((ext_vector_type(8)))  __bf16 v8bf;
typedef __attribute__((ext_vector_type(16))) __bf16 v16bf;
typedef __attribute__((ext_vector_type(8)))  float  v8f;

#define GEMM_BIAS  1
#define GEMM_RES   2
#define GEMM_GELU  4
#define GEMM_OUTBF 8

__device__ __forceinline__ unsigned short f2bf(float f) {
  unsigned int u = __float_as_uint(f);
  u += 0x7FFFu + ((u >> 16) & 1u);                 // round-to-nearest-even
  return (unsigned short)(u >> 16);
}
__device__ __forceinline__ float bf2f(unsigned short u) {
  return __uint_as_float(((unsigned)u) << 16);
}

// 16B global -> LDS async copy (ASYNCcnt-tracked, no VGPR data round-trip)
__device__ __forceinline__ void async_cp16(unsigned lds_off, const void* g) {
  asm volatile("global_load_async_to_lds_b128 %0, %1, off"
               :: "v"(lds_off), "v"(g) : "memory");
}

// ---------------------------------------------------------------------------
// bf16 WMMA GEMM: C[M,N] = A[M,K] @ Bt[N,K]^T (+bias) (+gelu) (+res)
// A, Bt are pre-converted bf16, both row-contiguous in K -> async LDS staging.
// Block tile 128x128, K-step 32, 8 waves, wave tile 32x64, 2-deep LDS pipeline.
// ---------------------------------------------------------------------------
__launch_bounds__(256)
__global__ void gemm_bf16_kernel(const unsigned short* __restrict__ A,
                                 const unsigned short* __restrict__ Bt,
                                 const float* __restrict__ bias,
                                 const float* __restrict__ res,
                                 void* __restrict__ Cout,
                                 int M, int N, int K, int flags) {
  constexpr int BM = 128, BN = 128, BK = 32, LDT = 40;   // 80B padded row pitch
  constexpr unsigned BUFB = BM * LDT * 2;                // 10240B per buffer
  __shared__ __align__(16) unsigned short As[2][BM * LDT];
  __shared__ __align__(16) unsigned short Bs[2][BN * LDT];

  const int tid  = threadIdx.x;
  const int lane = tid & 31;
  const int wave = tid >> 5;
  const int wm = (wave >> 1) * 32;
  const int wn = (wave & 1) * 64;
  const int bm = blockIdx.y * BM;
  const int bn = blockIdx.x * BN;

  // staging coords: 512 16B-chunks per operand tile; 2 chunks/thread/operand
  const int srow = tid >> 2;            // 0..63 (+64 for second chunk)
  const int sq   = tid & 3;             // 16B chunk within 64B row
  const unsigned uAs = (unsigned)(unsigned long long)&As[0][0];
  const unsigned uBs = (unsigned)(unsigned long long)&Bs[0][0];
  const unsigned lrow = (unsigned)(srow * LDT + sq * 8) * 2u;
  const unsigned short* aG = A  + (size_t)(bm + srow) * K + sq * 8;
  const unsigned short* bG = Bt + (size_t)(bn + srow) * K + sq * 8;

  auto issue = [&](int buf, int kt) {
    const unsigned short* a0 = aG + kt;
    const unsigned short* b0 = bG + kt;
    const unsigned la = uAs + (unsigned)buf * BUFB + lrow;
    const unsigned lb = uBs + (unsigned)buf * BUFB + lrow;
    async_cp16(la,         a0);
    async_cp16(la + 5120u, a0 + (size_t)64 * K);   // rows +64
    async_cp16(lb,         b0);
    async_cp16(lb + 5120u, b0 + (size_t)64 * K);
  };

  v8f acc[2][4] = {};
  const int mrow  = lane & 15;
  const int abase = (lane >> 4) * 8;    // A fragment K base
  const int khalf = (lane >> 4) * 8;    // B fragment K base

  const int nt = K / BK;
  issue(0, 0);
  for (int it = 0; it < nt; it++) {
    const int buf = it & 1;
    if (it + 1 < nt) {
      issue(buf ^ 1, (it + 1) * BK);
      asm volatile("s_wait_asynccnt 0x4" ::: "memory");  // tile 'it' landed
    } else {
      asm volatile("s_wait_asynccnt 0x0" ::: "memory");
    }
    __syncthreads();                    // all waves' copies for 'it' visible

    v16bf afr[2], bfr[4];
#pragma unroll
    for (int i = 0; i < 2; i++) {
      const int r = wm + i * 16 + mrow;
      const v8bf c0 = *(const v8bf*)&As[buf][r * LDT + abase];
      const v8bf c1 = *(const v8bf*)&As[buf][r * LDT + abase + 16];
#pragma unroll
      for (int e = 0; e < 8; e++) { afr[i][e] = c0[e]; afr[i][e + 8] = c1[e]; }
    }
#pragma unroll
    for (int j = 0; j < 4; j++) {
      const int n = wn + j * 16 + mrow;
      const v8bf c0 = *(const v8bf*)&Bs[buf][n * LDT + khalf];
      const v8bf c1 = *(const v8bf*)&Bs[buf][n * LDT + khalf + 16];
#pragma unroll
      for (int e = 0; e < 8; e++) { bfr[j][e] = c0[e]; bfr[j][e + 8] = c1[e]; }
    }
#pragma unroll
    for (int i = 0; i < 2; i++)
#pragma unroll
      for (int j = 0; j < 4; j++)
        acc[i][j] = __builtin_amdgcn_wmma_f32_16x16x32_bf16(
            false, afr[i], false, bfr[j], (short)0, acc[i][j], false, false);
    __syncthreads();                    // done reading buf before it is refilled
  }

  // epilogue: VGPR r holds (M = r + 8*(lane>=16), N = lane&15)
  const int nlane = lane & 15;
  const int rmoff = (lane >> 4) * 8;
#pragma unroll
  for (int i = 0; i < 2; i++) {
#pragma unroll
    for (int j = 0; j < 4; j++) {
      const int n = bn + wn + j * 16 + nlane;
      const float bv = (flags & GEMM_BIAS) ? bias[n] : 0.0f;
#pragma unroll
      for (int r = 0; r < 8; r++) {
        const int m = bm + wm + i * 16 + rmoff + r;
        float v = acc[i][j][r] + bv;
        if (flags & GEMM_GELU) v = 0.5f * v * (1.0f + erff(v * 0.70710678f));
        if (flags & GEMM_RES)  v += res[(size_t)m * N + n];
        if (flags & GEMM_OUTBF) ((unsigned short*)Cout)[(size_t)m * N + n] = f2bf(v);
        else                    ((float*)Cout)[(size_t)m * N + n] = v;
      }
    }
  }
}

// ---------------------------------------------------------------------------
// RMSNorm: bf16 out (feeds GEMM A operand)
// ---------------------------------------------------------------------------
__global__ void rmsnorm_kernel(const float* __restrict__ x,
                               const float* __restrict__ w,
                               unsigned short* __restrict__ y, int Dlen) {
  const int row = blockIdx.x;
  const float* xr = x + (size_t)row * Dlen;
  float ss = 0.f;
  for (int d = threadIdx.x; d < Dlen; d += blockDim.x) { float v = xr[d]; ss += v * v; }
  __shared__ float red[256];
  red[threadIdx.x] = ss;
  __syncthreads();
  for (int s = 128; s > 0; s >>= 1) {
    if ((int)threadIdx.x < s) red[threadIdx.x] += red[threadIdx.x + s];
    __syncthreads();
  }
  const float inv = 1.0f / fmaxf(sqrtf(red[0] / (float)Dlen), 1e-8f);
  unsigned short* yr = y + (size_t)row * Dlen;
  for (int d = threadIdx.x; d < Dlen; d += blockDim.x) yr[d] = f2bf(xr[d] * inv * w[d]);
}

// ---------------------------------------------------------------------------
// float -> bf16 bulk convert (chunked_seq)
// ---------------------------------------------------------------------------
__global__ void cvt_bf16_kernel(const float* __restrict__ s,
                                unsigned short* __restrict__ d, size_t n) {
  size_t i = (size_t)blockIdx.x * blockDim.x + threadIdx.x;
  const size_t stride = (size_t)gridDim.x * blockDim.x;
  for (; i < n; i += stride) d[i] = f2bf(s[i]);
}

// ---------------------------------------------------------------------------
// Weight transpose+convert: W[K,N] f32 -> Wt[N,K] bf16 (32x32 LDS tiles)
// ---------------------------------------------------------------------------
__global__ void wcvt_kernel(const float* __restrict__ W,
                            unsigned short* __restrict__ Wt, int Kd, int Nd) {
  __shared__ float tile[32][33];
  const int n0 = blockIdx.x * 32, k0 = blockIdx.y * 32;
  for (int r = threadIdx.y; r < 32; r += 8)
    tile[r][threadIdx.x] = W[(size_t)(k0 + r) * Nd + n0 + threadIdx.x];
  __syncthreads();
  for (int r = threadIdx.y; r < 32; r += 8)
    Wt[(size_t)(n0 + r) * Kd + k0 + threadIdx.x] = f2bf(tile[threadIdx.x][r]);
}

// ---------------------------------------------------------------------------
// Attention (bf16 in/out), RoPE-32 on q/k, exact softmax. ~5% of FLOPs.
// ---------------------------------------------------------------------------
__global__ void attn_kernel(const unsigned short* __restrict__ Q,
                            const unsigned short* __restrict__ Kg,
                            const unsigned short* __restrict__ Vg,
                            const unsigned char* __restrict__ mask,
                            unsigned short* __restrict__ O, int SQlen, int SKlen) {
  extern __shared__ float smem[];
  const int h = blockIdx.x, b = blockIdx.y;
  const int tid = threadIdx.x;
  const int ldk = 65;
  float* Ks   = smem;
  float* Vs   = Ks + (size_t)SKlen * ldk;
  float* qrow = Vs + (size_t)SKlen * ldk;
  float* sc   = qrow + 64;
  float* red  = sc + SKlen;

  const float LOG_STEP = 0.575646273249f;  // ln(10000)/16

  for (int idx = tid; idx < SKlen * 64; idx += blockDim.x) {
    const int j = idx >> 6, d = idx & 63;
    const size_t base = ((size_t)(b * SKlen + j)) * 1024 + h * 64;
    float kv = bf2f(Kg[base + d]);
    if (d < 32) {
      const int p = (d < 16) ? d + 16 : d - 16;
      const float partner = bf2f(Kg[base + p]);
      const float rot = (d < 16) ? -partner : partner;
      const float f = (float)j * expf(-(float)(d & 15) * LOG_STEP);
      kv = kv * cosf(f) + rot * sinf(f);
    }
    Ks[j * ldk + d] = kv;
    Vs[j * ldk + d] = bf2f(Vg[base + d]);
  }
  __syncthreads();

  const float qscale = 0.125f;  // DH^-0.5
  for (int i = 0; i < SQlen; i++) {
    const size_t qbase = ((size_t)(b * SQlen + i)) * 1024 + h * 64;
    if (tid < 64) {
      const int d = tid;
      float qv = bf2f(Q[qbase + d]) * qscale;
      if (d < 32) {
        const int p = (d < 16) ? d + 16 : d - 16;
        const float partner = bf2f(Q[qbase + p]) * qscale;
        const float rot = (d < 16) ? -partner : partner;
        const float f = (float)i * expf(-(float)(d & 15) * LOG_STEP);
        qv = qv * cosf(f) + rot * sinf(f);
      }
      qrow[d] = qv;
    }
    __syncthreads();

    float pmax = -3.4e38f;
    for (int j = tid; j < SKlen; j += blockDim.x) {
      const float* kr = Ks + j * ldk;
      float dot = 0.f;
#pragma unroll 16
      for (int d = 0; d < 64; d++) dot += qrow[d] * kr[d];
      if (mask && !mask[b * SQlen + j]) dot = -3.4e38f;
      sc[j] = dot;
      pmax = fmaxf(pmax, dot);
    }
    red[tid] = pmax; __syncthreads();
    for (int s = 128; s > 0; s >>= 1) {
      if (tid < s) red[tid] = fmaxf(red[tid], red[tid + s]);
      __syncthreads();
    }
    const float mval = red[0]; __syncthreads();

    float psum = 0.f;
    for (int j = tid; j < SKlen; j += blockDim.x) {
      const float p = expf(sc[j] - mval);
      sc[j] = p;
      psum += p;
    }
    red[tid] = psum; __syncthreads();
    for (int s = 128; s > 0; s >>= 1) {
      if (tid < s) red[tid] += red[tid + s];
      __syncthreads();
    }
    const float inv = 1.0f / red[0]; __syncthreads();

    const int dsub = tid & 63, part = tid >> 6;
    float acc = 0.f;
    for (int j = part; j < SKlen; j += 4) acc += sc[j] * Vs[j * ldk + dsub];
    red[tid] = acc; __syncthreads();
    if (tid < 64)
      O[qbase + tid] =
          f2bf((red[tid] + red[tid + 64] + red[tid + 128] + red[tid + 192]) * inv);
    __syncthreads();
  }
}

// ---------------------------------------------------------------------------
// Host orchestration
// ---------------------------------------------------------------------------
static inline void launch_gemm(const unsigned short* A, const unsigned short* Bt,
                               const float* bias, const float* res, void* C,
                               int M, int N, int K, int flags, hipStream_t s) {
  dim3 g(N / 128, M / 128);
  gemm_bf16_kernel<<<g, dim3(256), 0, s>>>(A, Bt, bias, res, C, M, N, K, flags);
}

extern "C" void kernel_launch(void* const* d_in, const int* in_sizes, int n_in,
                              void* d_out, int out_size, void* d_ws, size_t ws_size,
                              hipStream_t stream) {
  (void)in_sizes; (void)n_in; (void)out_size;
  constexpr int Dm = 1024, FFm = 4096, Bm = 64, SQm = 128, SKm = 512;
  constexpr int TQ = Bm * SQm;   // 8192 query tokens
  constexpr int TK = Bm * SKm;   // 32768 context tokens
  constexpr size_t MEG = 1024 * 1024;

  const float* x_in    = (const float*)d_in[0];
  const float* cs      = (const float*)d_in[1];
  const unsigned char* mask = (const unsigned char*)d_in[2];
  const float* attn_nw = (const float*)d_in[3];
  const float* Wq  = (const float*)d_in[4];
  const float* Wk  = (const float*)d_in[5];
  const float* Wv  = (const float*)d_in[6];
  const float* Wo  = (const float*)d_in[7];
  const float* bo  = (const float*)d_in[8];
  const float* cross_nw = (const float*)d_in[9];
  const float* cWq = (const float*)d_in[10];
  const float* cWk = (const float*)d_in[11];
  const float* cWv = (const float*)d_in[12];
  const float* cWo = (const float*)d_in[13];
  const float* cbo = (const float*)d_in[14];
  const float* ff_nw = (const float*)d_in[15];
  const float* ff_w1 = (const float*)d_in[16];
  const float* ff_b1 = (const float*)d_in[17];
  const float* ff_w2 = (const float*)d_in[18];
  const float* ff_b2 = (const float*)d_in[19];
  const float* out_nw = (const float*)d_in[20];
  const float* proj_w = (const float*)d_in[21];
  const float* proj_b = (const float*)d_in[22];

  // workspace carve-up
  char* p = (char*)d_ws;
  float*          X   = (float*)p;          p += (size_t)TQ * Dm * 4;    // residual (f32)
  unsigned short* H   = (unsigned short*)p; p += (size_t)TQ * Dm * 2;    // normed acts (bf16)
  unsigned short* Qb  = (unsigned short*)p; p += (size_t)TQ * Dm * 2;
  unsigned short* Kb  = (unsigned short*)p; p += (size_t)TK * Dm * 2;
  unsigned short* Vb  = (unsigned short*)p; p += (size_t)TK * Dm * 2;
  unsigned short* Tb  = (unsigned short*)p; p += (size_t)TQ * FFm * 2;   // FFN hidden / attn-out
  unsigned short* CSb = (unsigned short*)p; p += (size_t)TK * Dm * 2;    // chunked_seq bf16
  unsigned short* WT  = (unsigned short*)p; p += (size_t)33 * MEG * 2;   // transposed bf16 weights
  if ((size_t)(p - (char*)d_ws) > ws_size) return;
  unsigned short* AO = Tb;

  // transposed-weight table: per layer 16M elems:
  //   [0]=Wq [1]=Wk [2]=Wv [3]=Wo [4]=cWq [5]=cWk [6]=cWv [7]=cWo (1M each)
  //   [8M]=ff_w1(4M) [12M]=ff_w2(4M);  proj at 32M
  auto WTl = [&](int l, size_t off) { return WT + (size_t)l * 16 * MEG + off; };

  const int selfSmem  = (SQm * 65 * 2 + 64 + SQm + 256) * (int)sizeof(float);
  const int crossSmem = (SKm * 65 * 2 + 64 + SKm + 256) * (int)sizeof(float);
  hipFuncSetAttribute(reinterpret_cast<const void*>(attn_kernel),
                      hipFuncAttributeMaxDynamicSharedMemorySize, crossSmem);

  // one-time (per launch) conversions
  const dim3 tb(32, 8);
  for (int l = 0; l < 2; l++) {
    const size_t wDD = (size_t)l * MEG;
    wcvt_kernel<<<dim3(32, 32),  tb, 0, stream>>>(Wq  + wDD, WTl(l, 0 * MEG), Dm, Dm);
    wcvt_kernel<<<dim3(32, 32),  tb, 0, stream>>>(Wk  + wDD, WTl(l, 1 * MEG), Dm, Dm);
    wcvt_kernel<<<dim3(32, 32),  tb, 0, stream>>>(Wv  + wDD, WTl(l, 2 * MEG), Dm, Dm);
    wcvt_kernel<<<dim3(32, 32),  tb, 0, stream>>>(Wo  + wDD, WTl(l, 3 * MEG), Dm, Dm);
    wcvt_kernel<<<dim3(32, 32),  tb, 0, stream>>>(cWq + wDD, WTl(l, 4 * MEG), Dm, Dm);
    wcvt_kernel<<<dim3(32, 32),  tb, 0, stream>>>(cWk + wDD, WTl(l, 5 * MEG), Dm, Dm);
    wcvt_kernel<<<dim3(32, 32),  tb, 0, stream>>>(cWv + wDD, WTl(l, 6 * MEG), Dm, Dm);
    wcvt_kernel<<<dim3(32, 32),  tb, 0, stream>>>(cWo + wDD, WTl(l, 7 * MEG), Dm, Dm);
    wcvt_kernel<<<dim3(128, 32), tb, 0, stream>>>(ff_w1 + (size_t)l * 4 * MEG,
                                                  WTl(l, 8 * MEG), Dm, FFm);
    wcvt_kernel<<<dim3(32, 128), tb, 0, stream>>>(ff_w2 + (size_t)l * 4 * MEG,
                                                  WTl(l, 12 * MEG), FFm, Dm);
  }
  wcvt_kernel<<<dim3(32, 32), tb, 0, stream>>>(proj_w, WT + (size_t)32 * MEG, Dm, Dm);
  cvt_bf16_kernel<<<4096, 256, 0, stream>>>(cs, CSb, (size_t)TK * Dm);

  hipMemcpyAsync(X, x_in, (size_t)TQ * Dm * 4, hipMemcpyDeviceToDevice, stream);

  for (int l = 0; l < 2; l++) {
    // ---- self attention ----
    rmsnorm_kernel<<<TQ, 256, 0, stream>>>(X, attn_nw + l * Dm, H, Dm);
    launch_gemm(H, WTl(l, 0 * MEG), nullptr, nullptr, Qb, TQ, Dm, Dm, GEMM_OUTBF, stream);
    launch_gemm(H, WTl(l, 1 * MEG), nullptr, nullptr, Kb, TQ, Dm, Dm, GEMM_OUTBF, stream);
    launch_gemm(H, WTl(l, 2 * MEG), nullptr, nullptr, Vb, TQ, Dm, Dm, GEMM_OUTBF, stream);
    attn_kernel<<<dim3(16, Bm), 256, selfSmem, stream>>>(Qb, Kb, Vb, mask, AO, SQm, SQm);
    launch_gemm(AO, WTl(l, 3 * MEG), bo + l * Dm, X, X, TQ, Dm, Dm,
                GEMM_BIAS | GEMM_RES, stream);

    // ---- cross attention ----
    rmsnorm_kernel<<<TQ, 256, 0, stream>>>(X, cross_nw + l * Dm, H, Dm);
    launch_gemm(H,   WTl(l, 4 * MEG), nullptr, nullptr, Qb, TQ, Dm, Dm, GEMM_OUTBF, stream);
    launch_gemm(CSb, WTl(l, 5 * MEG), nullptr, nullptr, Kb, TK, Dm, Dm, GEMM_OUTBF, stream);
    launch_gemm(CSb, WTl(l, 6 * MEG), nullptr, nullptr, Vb, TK, Dm, Dm, GEMM_OUTBF, stream);
    attn_kernel<<<dim3(16, Bm), 256, crossSmem, stream>>>(Qb, Kb, Vb, nullptr, AO, SQm, SKm);
    launch_gemm(AO, WTl(l, 7 * MEG), cbo + l * Dm, X, X, TQ, Dm, Dm,
                GEMM_BIAS | GEMM_RES, stream);

    // ---- feed forward ----
    rmsnorm_kernel<<<TQ, 256, 0, stream>>>(X, ff_nw + l * Dm, H, Dm);
    launch_gemm(H, WTl(l, 8 * MEG), ff_b1 + (size_t)l * FFm, nullptr, Tb,
                TQ, FFm, Dm, GEMM_BIAS | GEMM_GELU | GEMM_OUTBF, stream);
    launch_gemm(Tb, WTl(l, 12 * MEG), ff_b2 + l * Dm, X, X,
                TQ, Dm, FFm, GEMM_BIAS | GEMM_RES, stream);
  }

  // ---- output norm + projection (f32 out) ----
  rmsnorm_kernel<<<TQ, 256, 0, stream>>>(X, out_nw, H, Dm);
  launch_gemm(H, WT + (size_t)32 * MEG, proj_b, nullptr, d_out, TQ, Dm, Dm,
              GEMM_BIAS, stream);
}
